// SlotSequenceEncoder_47364899340348
// MI455X (gfx1250) — matrix operand order repaired and checked by reading, not compiled, not run on hardware
//
#include <hip/hip_runtime.h>
#include <hip/hip_bf16.h>

// ---------------- problem constants ----------------
constexpr int kS  = 7;
constexpr int kB  = 4096;
constexpr int kD  = 1024;
constexpr int kE  = 128;
constexpr int kH  = 512;
constexpr int kV  = 200;
constexpr int kD0 = kD + kE;      // 1152
constexpr int kKA = kD0 + kH;     // 1664
constexpr int kFH = 4 * kH;       // 2048

constexpr int kBT = 32;           // batch rows per workgroup (2 M-tiles)
constexpr int kMT = kBT / 16;     // 2

// LDS row strides (halves / floats): multiple of 8 halves (16B loads) and
// stride/2 words % 64 != 0 (bank spread: 836%64=4, 516%64=4)
constexpr int kLdA = kKA + 8;     // 1672
constexpr int kLdH = 2 * kH + 8;  // 1032
constexpr int kLdC = kH + 4;      // 516

constexpr size_t kSmemBytes =
    (size_t)kBT * kLdA * 2 + (size_t)kBT * kLdH * 2 + 2 * (size_t)kBT * kLdC * 4;

// swizzled f16 weight arena offsets (halves)
constexpr size_t nWih0 = (size_t)kS * kFH * kD0;
constexpr size_t nWhh0 = (size_t)kS * kFH * kH;
constexpr size_t nHw0  = (size_t)kS * kH * kKA;
constexpr size_t nLin0 = (size_t)kS * kH * kD0;
constexpr size_t nWih1 = (size_t)kS * kFH * kH;
constexpr size_t nWhh1 = (size_t)kS * kFH * kH;
constexpr size_t oWih0 = 0;
constexpr size_t oWhh0 = oWih0 + nWih0;
constexpr size_t oHw0  = oWhh0 + nWhh0;
constexpr size_t oLin0 = oHw0 + nHw0;
constexpr size_t oWih1 = oLin0 + nLin0;
constexpr size_t oWhh1 = oWih1 + nWih1;
constexpr size_t kWsHalves = oWhh1 + nWhh1;          // 48,627,712
constexpr size_t kWsBytes  = kWsHalves * 2;          // ~92.8 MiB

typedef _Float16 v16h __attribute__((ext_vector_type(16)));
typedef _Float16 v8h  __attribute__((ext_vector_type(8)));
typedef float    v8f  __attribute__((ext_vector_type(8)));

__device__ __forceinline__ v8f zero8() {
  v8f z = {0.f, 0.f, 0.f, 0.f, 0.f, 0.f, 0.f, 0.f};
  return z;
}
__device__ __forceinline__ float sigm(float x) {
  return 1.0f / (1.0f + __expf(-x));
}
__device__ __forceinline__ float tanh_fast(float x) {
  float e = __expf(-2.0f * fabsf(x));
  float r = (1.0f - e) / (1.0f + e);
  return (x < 0.0f) ? -r : r;
}

// A fragment (16x32 f16) from LDS, contiguous k order (B uses the same):
// lane<16: K=kBase+{0..15}; lane>=16: K=kBase+{16..31}
__device__ __forceinline__ v16h load_a_frag(const _Float16* row0, int lda,
                                            int kBase, int lane) {
  const int m    = lane & 15;
  const int kOff = (lane < 16) ? 0 : 16;
  const _Float16* p = row0 + m * lda + kBase + kOff;
  v8h a0 = *(const v8h*)(p);
  v8h a1 = *(const v8h*)(p + 8);
  v16h A;
#pragma unroll
  for (int i = 0; i < 8; ++i) { A[i] = a0[i]; A[8 + i] = a1[i]; }
  return A;
}

template <int NM>
__device__ __forceinline__ void load_a(v16h (&A)[NM], const _Float16* buf, int lda,
                                       int kBase, int lane) {
#pragma unroll
  for (int mt = 0; mt < NM; ++mt)
    A[mt] = load_a_frag(buf + mt * 16 * lda, lda, kBase, lane);
}

// ---- B fragment loaders (overloaded on weight type) ----
// f32 fallback: gather from [N,K] row-major + cvt to f16 pairs.
__device__ __forceinline__ v16h load_b(const float* W, int ldb, int nBase,
                                       int kBase, int lane) {
  const int n = nBase + (lane & 15);
  const int k = kBase + ((lane < 16) ? 0 : 16);
  const float* p = W + (size_t)n * ldb + k;
  v16h Bv;
#pragma unroll
  for (int g = 0; g < 8; ++g) {
    float2 w = *(const float2*)(p + 2 * g);
    Bv[2 * g]     = (_Float16)w.x;
    Bv[2 * g + 1] = (_Float16)w.y;
  }
  return Bv;
}
// f16 pre-swizzled arena: block (ntile,kchunk) = 512 halves; lane l owns
// halves [l*16, l*16+16) = exactly its WMMA B operand. One 32B coalesced load.
__device__ __forceinline__ v16h load_b(const _Float16* W, int ldb, int nBase,
                                       int kBase, int lane) {
  const int kChunks = ldb >> 5;
  const size_t blk = (size_t)(nBase >> 4) * kChunks + (kBase >> 5);
  return *(const v16h*)(W + blk * 512 + lane * 16);
}

__device__ __forceinline__ v8f wmma_f16(const v16h& A, const v16h& B, const v8f& C) {
  return __builtin_amdgcn_wmma_f32_16x16x32_f16(false, A, false, B, (short)0, C,
                                                false, false);
}

// One K-step: NQ resident B fragments, each reused across NM M-tiles.
template <int NQ, int NM, typename WT>
__device__ __forceinline__ void gemm_step(v8f (&acc)[NQ][NM], const v16h (&A)[NM],
                                          const WT* W, int ldb, int kw,
                                          int nBase0, int qStride, int lane) {
  v16h Bf[NQ];
#pragma unroll
  for (int q = 0; q < NQ; ++q)
    Bf[q] = load_b(W, ldb, nBase0 + q * qStride, kw, lane);
#pragma unroll
  for (int q = 0; q < NQ; ++q)
#pragma unroll
    for (int mt = 0; mt < NM; ++mt)
      acc[q][mt] = wmma_f16(A[mt], Bf[q], acc[q][mt]);
}

// One LSTM layer: gates GEMM over [xPart|hPart] + elementwise; h staged in regs
// (written to LDS by the caller after a barrier), c-state updated in LDS.
template <typename WT>
__device__ __forceinline__ void lstm_layer(const _Float16* aBuf, int lda, int lenX,
                                           const WT* Wx, const WT* Wh,
                                           const float* bih, const float* bhh,
                                           int sOff, float* lC, v8f (&hreg)[4][kMT],
                                           int wave, int lane, int laneN, int rowHi) {
#pragma unroll 1
  for (int t = 0; t < 4; ++t) {
    const int nb0 = wave * 64 + t * 16;
    v8f acc[4][kMT];
#pragma unroll
    for (int q = 0; q < 4; ++q)
#pragma unroll
      for (int mt = 0; mt < kMT; ++mt) acc[q][mt] = zero8();

    for (int ks = 0; ks < lenX / 32; ++ks) {
      v16h A[kMT];
      load_a(A, aBuf, lda, ks * 32, lane);
      gemm_step<4, kMT>(acc, A, Wx, lenX, ks * 32, nb0, kH, lane);
    }
    for (int ks = 0; ks < kH / 32; ++ks) {
      v16h A[kMT];
      load_a(A, aBuf, lda, lenX + ks * 32, lane);
      gemm_step<4, kMT>(acc, A, Wh, kH, ks * 32, nb0, kH, lane);
    }

    const int j  = nb0 + laneN;
    const float bi = bih[sOff + j]          + bhh[sOff + j];
    const float bf = bih[sOff + kH + j]     + bhh[sOff + kH + j];
    const float bg = bih[sOff + 2 * kH + j] + bhh[sOff + 2 * kH + j];
    const float bo = bih[sOff + 3 * kH + j] + bhh[sOff + 3 * kH + j];
#pragma unroll
    for (int mt = 0; mt < kMT; ++mt)
#pragma unroll
      for (int g = 0; g < 8; ++g) {
        const int m = mt * 16 + g + rowHi;
        const float iv = sigm(acc[0][mt][g] + bi);
        const float fv = sigm(acc[1][mt][g] + bf);
        const float gv = tanh_fast(acc[2][mt][g] + bg);
        const float ov = sigm(acc[3][mt][g] + bo);
        const float cn = fv * lC[m * kLdC + j] + iv * gv;
        lC[m * kLdC + j] = cn;
        hreg[t][mt][g] = ov * tanh_fast(cn);
      }
  }
}

template <typename WT>
__global__ __launch_bounds__(256, 1) void slot_seq_encoder(
    const float* __restrict__ pred,    // [B, D]
    const int*   __restrict__ labels,  // [S, B]
    const float* __restrict__ emb,     // [S, V, E]
    const WT* __restrict__ Wih0, const WT* __restrict__ Whh0,
    const float* __restrict__ bih0, const float* __restrict__ bhh0,
    const WT* __restrict__ hwW0, const float* __restrict__ hwb0,
    const WT* __restrict__ linW0,
    const WT* __restrict__ Wih1, const WT* __restrict__ Whh1,
    const float* __restrict__ bih1, const float* __restrict__ bhh1,
    float* __restrict__ out)           // [B, H]
{
  extern __shared__ char smemRaw[];
  _Float16* lA  = (_Float16*)smemRaw;                 // [x | h0]   f16
  _Float16* lH1 = lA + (size_t)kBT * kLdA;            // [x1 | h1]  f16
  float*    lC0 = (float*)(lH1 + (size_t)kBT * kLdH); // c0 state
  float*    lC1 = lC0 + (size_t)kBT * kLdC;           // c1 state

  const int tid   = threadIdx.x;
  const int lane  = tid & 31;
  const int wave  = tid >> 5;
  const int R     = blockIdx.x * kBT;
  const int laneN = lane & 15;
  const int rowHi = (lane < 16) ? 0 : 8;

  // one-time staging: pred (constant across slots), zero h/c state
  for (int idx = tid; idx < kBT * kD; idx += 256) {
    const int m = idx >> 10, k = idx & (kD - 1);
    lA[m * kLdA + k] = (_Float16)pred[(size_t)(R + m) * kD + k];
  }
  for (int idx = tid; idx < kBT * kH; idx += 256) {
    const int m = idx >> 9, j = idx & (kH - 1);
    lA[m * kLdA + kD0 + j] = (_Float16)0.0f;
    lH1[m * kLdH + kH + j] = (_Float16)0.0f;
    lC0[m * kLdC + j] = 0.0f;
    lC1[m * kLdC + j] = 0.0f;
  }
  __syncthreads();

  for (int s = 0; s < kS; ++s) {
    // Phase A: embedding gather into x[:, 1024:1152]
    for (int idx = tid; idx < kBT * kE; idx += 256) {
      const int m = idx >> 7, k = idx & (kE - 1);
      const int lbl = labels[s * kB + R + m];
      lA[m * kLdA + kD + k] = (_Float16)emb[((size_t)s * kV + lbl) * kE + k];
    }
    __syncthreads();

    // Phase B/C: layer-0 LSTM (gates GEMM + elementwise), delayed h0 write
    v8f hreg[4][kMT];
    lstm_layer(lA, kLdA, kD0,
               Wih0 + (size_t)s * kFH * kD0, Whh0 + (size_t)s * kFH * kH,
               bih0, bhh0, s * kFH, lC0, hreg, wave, lane, laneN, rowHi);
    __syncthreads();  // all waves done reading old h0
#pragma unroll
    for (int t = 0; t < 4; ++t) {
      const int j = wave * 64 + t * 16 + laneN;
#pragma unroll
      for (int mt = 0; mt < kMT; ++mt)
#pragma unroll
        for (int g = 0; g < 8; ++g)
          lA[(mt * 16 + g + rowHi) * kLdA + kD0 + j] = (_Float16)hreg[t][mt][g];
    }
    __syncthreads();

    // Phase D/E: highway pre-act + linear path; x1 -> lH1 immediately per t
    {
      const WT* hwW = hwW0 + (size_t)s * kH * kKA;
      const WT* lnW = linW0 + (size_t)s * kH * kD0;
#pragma unroll 1
      for (int t = 0; t < 4; ++t) {
        const int nb0 = wave * 64 + t * 16;
        v8f accG[1][kMT], accL[1][kMT];
#pragma unroll
        for (int mt = 0; mt < kMT; ++mt) { accG[0][mt] = zero8(); accL[0][mt] = zero8(); }
        for (int ks = 0; ks < kD0 / 32; ++ks) {
          v16h A[kMT];
          load_a(A, lA, kLdA, ks * 32, lane);
          gemm_step<1, kMT>(accG, A, hwW, kKA, ks * 32, nb0, 0, lane);
          gemm_step<1, kMT>(accL, A, lnW, kD0, ks * 32, nb0, 0, lane);
        }
        for (int ks = 0; ks < kH / 32; ++ks) {
          v16h A[kMT];
          load_a(A, lA, kLdA, kD0 + ks * 32, lane);
          gemm_step<1, kMT>(accG, A, hwW, kKA, kD0 + ks * 32, nb0, 0, lane);
        }
        const int j  = nb0 + laneN;
        const float bw = hwb0[s * kH + j];
#pragma unroll
        for (int mt = 0; mt < kMT; ++mt)
#pragma unroll
          for (int g = 0; g < 8; ++g) {
            const int m = mt * 16 + g + rowHi;
            const float gt  = sigm(accG[0][mt][g] + bw);
            const float h0n = (float)lA[m * kLdA + kD0 + j];
            const float x1  = gt * h0n + (1.0f - gt) * accL[0][mt][g];
            lH1[m * kLdH + j] = (_Float16)x1;
          }
      }
    }
    __syncthreads();

    // Phase F/G: layer-1 LSTM; delayed h1 write (+ output on last slot)
    v8f hreg1[4][kMT];
    lstm_layer(lH1, kLdH, kH,
               Wih1 + (size_t)s * kFH * kH, Whh1 + (size_t)s * kFH * kH,
               bih1, bhh1, s * kFH, lC1, hreg1, wave, lane, laneN, rowHi);
    __syncthreads();  // all waves done reading old h1
#pragma unroll
    for (int t = 0; t < 4; ++t) {
      const int j = wave * 64 + t * 16 + laneN;
#pragma unroll
      for (int mt = 0; mt < kMT; ++mt)
#pragma unroll
        for (int g = 0; g < 8; ++g) {
          const int m = mt * 16 + g + rowHi;
          lH1[m * kLdH + kH + j] = (_Float16)hreg1[t][mt][g];
          if (s == kS - 1)
            out[(size_t)(R + m) * kH + j] = hreg1[t][mt][g];
        }
    }
    __syncthreads();
  }
}

// f32 [S,N,K] -> f16 arena pre-swizzled into WMMA B-fragment blocks:
// block (ntile,kc) = 512 halves; element = lane*16 + h with
// n = ntile*16 + (lane&15), k = kc*32 + (lane<16?0:16) + h.
__global__ void convert_swizzle(const float* __restrict__ src,
                                _Float16* __restrict__ dst, int N, int K,
                                int nSlots) {
  const int kChunks = K >> 5;
  const size_t perSlot = (size_t)N * K;
  const size_t total = perSlot * nSlots;
  for (size_t i = (size_t)blockIdx.x * blockDim.x + threadIdx.x; i < total;
       i += (size_t)gridDim.x * blockDim.x) {
    const size_t sIdx = i / perSlot;
    const size_t r = i - sIdx * perSlot;
    const int blk  = (int)(r >> 9);
    const int e    = (int)(r & 511);
    const int lane = e >> 4, hh = e & 15;
    const int ntile = blk / kChunks, kc = blk - ntile * kChunks;
    const int n = ntile * 16 + (lane & 15);
    const int k = kc * 32 + ((lane < 16) ? 0 : 16) + hh;
    dst[i] = (_Float16)src[(sIdx * N + n) * (size_t)K + k];
  }
}

extern "C" void kernel_launch(void* const* d_in, const int* in_sizes, int n_in,
                              void* d_out, int out_size, void* d_ws, size_t ws_size,
                              hipStream_t stream) {
  (void)in_sizes; (void)n_in; (void)out_size;
  const float* pred   = (const float*)d_in[0];
  const int*   labels = (const int*)d_in[1];
  const float* emb    = (const float*)d_in[2];
  const float* Wih0   = (const float*)d_in[3];
  const float* Whh0   = (const float*)d_in[4];
  const float* bih0   = (const float*)d_in[5];
  const float* bhh0   = (const float*)d_in[6];
  const float* hwW0   = (const float*)d_in[7];
  const float* hwb0   = (const float*)d_in[8];
  const float* linW0  = (const float*)d_in[9];
  const float* Wih1   = (const float*)d_in[10];
  const float* Whh1   = (const float*)d_in[11];
  const float* bih1   = (const float*)d_in[12];
  const float* bhh1   = (const float*)d_in[13];
  float* out = (float*)d_out;

  const int grid = kB / kBT;  // 128 workgroups x 256 threads

  if (ws_size >= kWsBytes) {
    _Float16* w = (_Float16*)d_ws;
    convert_swizzle<<<2048, 256, 0, stream>>>(Wih0,  w + oWih0, kFH, kD0, kS);
    convert_swizzle<<<2048, 256, 0, stream>>>(Whh0,  w + oWhh0, kFH, kH,  kS);
    convert_swizzle<<<1024, 256, 0, stream>>>(hwW0,  w + oHw0,  kH,  kKA, kS);
    convert_swizzle<<<1024, 256, 0, stream>>>(linW0, w + oLin0, kH,  kD0, kS);
    convert_swizzle<<<2048, 256, 0, stream>>>(Wih1,  w + oWih1, kFH, kH,  kS);
    convert_swizzle<<<2048, 256, 0, stream>>>(Whh1,  w + oWhh1, kFH, kH,  kS);
    slot_seq_encoder<_Float16><<<grid, 256, kSmemBytes, stream>>>(
        pred, labels, emb, w + oWih0, w + oWhh0, bih0, bhh0, w + oHw0, hwb0,
        w + oLin0, w + oWih1, w + oWhh1, bih1, bhh1, out);
  } else {
    slot_seq_encoder<float><<<grid, 256, kSmemBytes, stream>>>(
        pred, labels, emb, Wih0, Whh0, bih0, bhh0, hwW0, hwb0, linW0, Wih1,
        Whh1, bih1, bhh1, out);
  }
}
